// SymbolicReasoningLayer_37864431681863
// MI455X (gfx1250) — compile-verified, matrix operand order checked by reading, not run on hardware
//
#include <hip/hip_runtime.h>

// ---------------------------------------------------------------------------
// Problem constants (match the reference)
// ---------------------------------------------------------------------------
#define Bsz    4
#define Ssz    1024
#define Hsz    1024
#define Rsz    512
#define Msz    1024
#define NHEADS 16
#define DHEAD  64
#define NSTEPS 3

typedef __attribute__((ext_vector_type(16))) __bf16 bf16x16;
typedef __attribute__((ext_vector_type(8)))  __bf16 bf16x8;
typedef __attribute__((ext_vector_type(8)))  float  f32x8;

// ---------------------------------------------------------------------------
// bf16 <-> f32 helpers (bit-level, no reliance on __bf16 arithmetic support)
// ---------------------------------------------------------------------------
__device__ __forceinline__ float bf2f(__bf16 x) {
  unsigned short s = __builtin_bit_cast(unsigned short, x);
  unsigned int u = ((unsigned int)s) << 16;
  return __builtin_bit_cast(float, u);
}
__device__ __forceinline__ __bf16 f2bf(float f) {
  unsigned int u = __builtin_bit_cast(unsigned int, f);
  u += 0x7fffu + ((u >> 16) & 1u);               // round-to-nearest-even
  unsigned short s = (unsigned short)(u >> 16);
  return __builtin_bit_cast(__bf16, s);
}

// ---------------------------------------------------------------------------
// VALU-only 16-lane reductions (rows of the WMMA C-layout live in 16-lane
// halves). xor1/2/4 stay inside an 8-lane DPP8 group; xor8 == row_ror:8 on a
// 16-lane DPP row. Falls back to __shfl_xor (ds_bpermute) if DPP builtins
// are unavailable.
// ---------------------------------------------------------------------------
#if __has_builtin(__builtin_amdgcn_mov_dpp8) && \
    __has_builtin(__builtin_amdgcn_update_dpp)
#define USE_DPP_RED 1
#endif

#ifdef USE_DPP_RED
constexpr int dpp8imm(int a, int b, int c, int d, int e, int f, int g, int h) {
  return a | (b << 3) | (c << 6) | (d << 9) | (e << 12) | (f << 15) |
         (g << 18) | (h << 21);
}
template <int IMM>
__device__ __forceinline__ float dpp8f(float v) {
  return __builtin_bit_cast(
      float, __builtin_amdgcn_mov_dpp8(__builtin_bit_cast(int, v), IMM));
}
__device__ __forceinline__ float ror8f(float v) {  // lane ^ 8 within DPP row
  int i = __builtin_bit_cast(int, v);
  return __builtin_bit_cast(
      float, __builtin_amdgcn_update_dpp(i, i, 0x128, 0xf, 0xf, true));
}
__device__ __forceinline__ float red16_max(float v) {
  v = fmaxf(v, dpp8f<dpp8imm(1, 0, 3, 2, 5, 4, 7, 6)>(v));  // xor 1
  v = fmaxf(v, dpp8f<dpp8imm(2, 3, 0, 1, 6, 7, 4, 5)>(v));  // xor 2
  v = fmaxf(v, dpp8f<dpp8imm(4, 5, 6, 7, 0, 1, 2, 3)>(v));  // xor 4
  v = fmaxf(v, ror8f(v));                                   // xor 8
  return v;
}
__device__ __forceinline__ float red16_sum(float v) {
  v += dpp8f<dpp8imm(1, 0, 3, 2, 5, 4, 7, 6)>(v);
  v += dpp8f<dpp8imm(2, 3, 0, 1, 6, 7, 4, 5)>(v);
  v += dpp8f<dpp8imm(4, 5, 6, 7, 0, 1, 2, 3)>(v);
  v += ror8f(v);
  return v;
}
#else
__device__ __forceinline__ float red16_max(float v) {
  for (int off = 8; off >= 1; off >>= 1) v = fmaxf(v, __shfl_xor(v, off));
  return v;
}
__device__ __forceinline__ float red16_sum(float v) {
  for (int off = 8; off >= 1; off >>= 1) v += __shfl_xor(v, off);
  return v;
}
#endif

// ---------------------------------------------------------------------------
// Load one 16x32 bf16 WMMA operand fragment (A-layout; also used for B since
// B operands are stored column-major == rows of the N x K transposed weight).
// ISA layout (CDNA5 ISA): lane m (m = lane&15) holds row m; lanes 0-15 take
// K = 0..7 & 16..23, lanes 16-31 take K = 8..15 & 24..31 -> two 16B chunks.
// ---------------------------------------------------------------------------
__device__ __forceinline__ bf16x16 frag_ld(const __bf16* base, int stride,
                                           int koff) {
  const int lane = threadIdx.x & 31;
  const __bf16* p = base + (lane & 15) * stride + koff + ((lane >> 4) << 3);
  union { bf16x16 v; bf16x8 h[2]; } u;
  u.h[0] = *(const bf16x8*)(p);
  u.h[1] = *(const bf16x8*)(p + 16);
  return u.v;
}

__device__ __forceinline__ f32x8 wmma_bf16(bf16x16 a, bf16x16 b, f32x8 c) {
  return __builtin_amdgcn_wmma_f32_16x16x32_bf16(false, a, false, b, (short)0,
                                                 c, false, false);
}

// ---------------------------------------------------------------------------
// Generic tiled bf16 WMMA GEMM:  out = A[M,K] @ Bt[N,K]^T (+bias)(+residual)
// Block tile 128x128, BK=32, 256 threads = 8 waves (4 along M x 2 along N),
// each wave computes 32x64 via 2x4 WMMA accumulators. Double-buffered LDS
// staging through registers: global loads for tile kt+1 are issued before the
// WMMAs on tile kt; one barrier per K-tile.
// ---------------------------------------------------------------------------
#define BM  128
#define BN  128
#define BK  32
#define SKA 48   // padded LDS K-stride (96B, 16B-aligned rows)

template <bool WF32, bool WBF16, bool RES>
__global__ __launch_bounds__(256) void gemm_bf16_wmma(
    const __bf16* __restrict__ A, int lda,
    const __bf16* __restrict__ Bt, int ldb,
    const float* __restrict__ bias,
    const float* __restrict__ resid, int ldr,
    float* __restrict__ outF, int ldof,
    __bf16* __restrict__ outB, int ldob,
    int Kdim) {
  __shared__ __attribute__((aligned(16))) __bf16 sA[2][BM * SKA];
  __shared__ __attribute__((aligned(16))) __bf16 sB[2][BN * SKA];

  const int tid  = threadIdx.x;
  const int lane = tid & 31;
  const int wave = tid >> 5;
  const int wm   = wave & 3;   // 0..3 -> M offset wm*32
  const int wn   = wave >> 2;  // 0..1 -> N offset wn*64
  const int mbase = blockIdx.y * BM;
  const int nbase = blockIdx.x * BN;

  // per-thread staging coordinates: chunks c = tid, tid+256 of 512 (16B each)
  const int rowS0 = tid >> 2,          colS0 = (tid & 3) << 3;
  const int rowS1 = (tid + 256) >> 2,  colS1 = ((tid + 256) & 3) << 3;

  f32x8 acc[2][4];
#pragma unroll
  for (int t = 0; t < 2; ++t)
#pragma unroll
    for (int n = 0; n < 4; ++n)
#pragma unroll
      for (int e = 0; e < 8; ++e) acc[t][n][e] = 0.f;

  const int kTiles = Kdim / BK;

  // prologue: stage K-tile 0 into buffer 0
  {
    *(bf16x8*)&sA[0][rowS0 * SKA + colS0] =
        *(const bf16x8*)(A + (size_t)(mbase + rowS0) * lda + colS0);
    *(bf16x8*)&sA[0][rowS1 * SKA + colS1] =
        *(const bf16x8*)(A + (size_t)(mbase + rowS1) * lda + colS1);
    *(bf16x8*)&sB[0][rowS0 * SKA + colS0] =
        *(const bf16x8*)(Bt + (size_t)(nbase + rowS0) * ldb + colS0);
    *(bf16x8*)&sB[0][rowS1 * SKA + colS1] =
        *(const bf16x8*)(Bt + (size_t)(nbase + rowS1) * ldb + colS1);
  }
  __syncthreads();

  for (int kt = 0; kt < kTiles; ++kt) {
    const int cur = kt & 1;
    const bool have = (kt + 1 < kTiles);

    // issue next tile's global loads first (latency hidden under WMMAs)
    bf16x8 ra0{}, ra1{}, rb0{}, rb1{};
    if (have) {
      const int k1 = (kt + 1) * BK;
      const __bf16* ga0 = A + (size_t)(mbase + rowS0) * lda + k1 + colS0;
      const __bf16* ga1 = A + (size_t)(mbase + rowS1) * lda + k1 + colS1;
      const __bf16* gb0 = Bt + (size_t)(nbase + rowS0) * ldb + k1 + colS0;
      const __bf16* gb1 = Bt + (size_t)(nbase + rowS1) * ldb + k1 + colS1;
      ra0 = *(const bf16x8*)ga0;
      ra1 = *(const bf16x8*)ga1;
      rb0 = *(const bf16x8*)gb0;
      rb1 = *(const bf16x8*)gb1;
      if (kt + 2 < kTiles) {           // pull tile kt+2 toward the caches
        __builtin_prefetch(ga0 + BK, 0, 1);
        __builtin_prefetch(gb0 + BK, 0, 1);
      }
    }

    // compute on current buffer
    bf16x16 af[2], bfr[4];
#pragma unroll
    for (int t = 0; t < 2; ++t)
      af[t] = frag_ld(&sA[cur][(wm * 32 + t * 16) * SKA], SKA, 0);
#pragma unroll
    for (int n = 0; n < 4; ++n)
      bfr[n] = frag_ld(&sB[cur][(wn * 64 + n * 16) * SKA], SKA, 0);
#pragma unroll
    for (int t = 0; t < 2; ++t)
#pragma unroll
      for (int n = 0; n < 4; ++n)
        acc[t][n] = wmma_bf16(af[t], bfr[n], acc[t][n]);

    // store prefetched tile into the other buffer
    if (have) {
      *(bf16x8*)&sA[cur ^ 1][rowS0 * SKA + colS0] = ra0;
      *(bf16x8*)&sA[cur ^ 1][rowS1 * SKA + colS1] = ra1;
      *(bf16x8*)&sB[cur ^ 1][rowS0 * SKA + colS0] = rb0;
      *(bf16x8*)&sB[cur ^ 1][rowS1 * SKA + colS1] = rb1;
    }
    __syncthreads();
  }

  // Epilogue. C-layout: VGPR r -> row = r + 8*(lane>=16), col = lane&15.
  const int colIn = lane & 15;
  const int rhalf = (lane >> 4) << 3;
#pragma unroll
  for (int t = 0; t < 2; ++t)
#pragma unroll
    for (int n = 0; n < 4; ++n)
#pragma unroll
      for (int r = 0; r < 8; ++r) {
        int row = mbase + wm * 32 + t * 16 + rhalf + r;
        int col = nbase + wn * 64 + n * 16 + colIn;
        float v = acc[t][n][r];
        if (bias) v += bias[col];
        if (RES) v += resid[(size_t)row * ldr + col];
        if (WF32) outF[(size_t)row * ldof + col] = v;
        if (WBF16) outB[(size_t)row * ldob + col] = f2bf(v);
      }
}

// ---------------------------------------------------------------------------
// Flash attention (d=64, kv_len=1024, scale=1/8). One block = 128 q-rows of
// one (batch, head); each wave owns 16 q-rows. K tile (32x64) and V tile
// (transposed, 64x32) staged in double-buffered LDS shared by all 8 waves.
// Online softmax in f32 with DPP row reductions; P round-trips through
// per-wave LDS to become a WMMA A-operand (guarded by s_wait_dscnt).
// ---------------------------------------------------------------------------
#define SK 72   // sK row stride (rows = kv, 64 d-cols), 144B rows
#define SV 40   // sVt row stride (rows = d, 32 kv-cols), 80B rows
#define SP 40   // per-wave P tile stride

__global__ __launch_bounds__(256) void flash_attn_bf16(
    const __bf16* __restrict__ Q, const __bf16* __restrict__ Kp,
    const __bf16* __restrict__ Vp, __bf16* __restrict__ O,
    long q_bstride, long kv_bstride, int kv_len, int ldk, float scale) {
  __shared__ __attribute__((aligned(16))) __bf16 sK[2][32 * SK];
  __shared__ __attribute__((aligned(16))) __bf16 sVt[2][64 * SV];
  __shared__ __attribute__((aligned(16))) __bf16 sP[8 * 16 * SP];

  const int tid  = threadIdx.x;
  const int lane = tid & 31;
  const int wave = tid >> 5;
  const int qtile = blockIdx.x * 128;
  const int head  = blockIdx.y;
  const int b     = blockIdx.z;

  // Q fragments for this wave's 16 rows (loaded once, straight from global)
  const size_t qoff =
      (size_t)b * q_bstride + (size_t)(qtile + wave * 16) * Hsz + head * DHEAD;
  bf16x16 qf[2];
  qf[0] = frag_ld(Q + qoff, Hsz, 0);
  qf[1] = frag_ld(Q + qoff, Hsz, 32);

  f32x8 o[4];
  float mi[8], li[8];
#pragma unroll
  for (int n = 0; n < 4; ++n)
#pragma unroll
    for (int e = 0; e < 8; ++e) o[n][e] = 0.f;
#pragma unroll
  for (int r = 0; r < 8; ++r) { mi[r] = -3.0e38f; li[r] = 0.f; }

  const size_t kvoff = (size_t)b * kv_bstride + head * DHEAD;
  __bf16* sPw = &sP[wave * 16 * SP];
  const int srow = tid >> 3;          // 0..31 kv row (staging)
  const int scc  = (tid & 7) << 3;    // 0..56 d chunk (staging)

  const int nT = kv_len / 32;

  // prologue: stage kv-tile 0 into buffer 0
  {
    bf16x8 kr = *(const bf16x8*)(Kp + kvoff + (size_t)srow * ldk + scc);
    bf16x8 vr = *(const bf16x8*)(Vp + kvoff + (size_t)srow * ldk + scc);
    *(bf16x8*)&sK[0][srow * SK + scc] = kr;
#pragma unroll
    for (int i = 0; i < 8; ++i) sVt[0][(scc + i) * SV + srow] = vr[i];
  }
  __syncthreads();

  for (int it = 0; it < nT; ++it) {
    const int cur = it & 1;
    const bool have = (it + 1 < nT);

    // issue next tile's global loads first
    bf16x8 kr{}, vr{};
    if (have) {
      const size_t roff = kvoff + (size_t)((it + 1) * 32 + srow) * ldk + scc;
      kr = *(const bf16x8*)(Kp + roff);
      vr = *(const bf16x8*)(Vp + roff);
    }

    // scores: 16x32 = two 16x16 C-tiles, accumulate over d=64 (2 k-steps)
    f32x8 sc[2];
#pragma unroll
    for (int nt = 0; nt < 2; ++nt) {
#pragma unroll
      for (int e = 0; e < 8; ++e) sc[nt][e] = 0.f;
#pragma unroll
      for (int kk = 0; kk < 2; ++kk) {
        bf16x16 kf = frag_ld(&sK[cur][nt * 16 * SK], SK, kk * 32);
        sc[nt] = wmma_bf16(qf[kk], kf, sc[nt]);
      }
    }

    // online softmax per row; VALU-only DPP reductions inside 16-lane halves
#pragma unroll
    for (int r = 0; r < 8; ++r) {
      float s0 = sc[0][r] * scale, s1 = sc[1][r] * scale;
      float mx = red16_max(fmaxf(s0, s1));
      float mnew = fmaxf(mi[r], mx);
      float p0 = __expf(s0 - mnew), p1 = __expf(s1 - mnew);
      float rs = red16_sum(p0 + p1);
      float alpha = __expf(mi[r] - mnew);
      li[r] = li[r] * alpha + rs;
      mi[r] = mnew;
#pragma unroll
      for (int nt = 0; nt < 4; ++nt) o[nt][r] *= alpha;
      int prow = ((lane >> 4) << 3) + r;
      sPw[prow * SP + (lane & 15)]      = f2bf(p0);
      sPw[prow * SP + 16 + (lane & 15)] = f2bf(p1);
    }
    asm volatile("s_wait_dscnt 0" ::: "memory");  // P stores -> P frag loads

    bf16x16 pf = frag_ld(sPw, SP, 0);
#pragma unroll
    for (int nt = 0; nt < 4; ++nt) {
      bf16x16 vf = frag_ld(&sVt[cur][nt * 16 * SV], SV, 0);
      o[nt] = wmma_bf16(pf, vf, o[nt]);
    }

    // store prefetched tile into the other buffer
    if (have) {
      *(bf16x8*)&sK[cur ^ 1][srow * SK + scc] = kr;
#pragma unroll
      for (int i = 0; i < 8; ++i) sVt[cur ^ 1][(scc + i) * SV + srow] = vr[i];
    }
    __syncthreads();
  }

  // normalize and write bf16 output (same [.,H] layout as Q)
#pragma unroll
  for (int r = 0; r < 8; ++r) {
    float inv = 1.f / li[r];
    int qrow = qtile + wave * 16 + ((lane >> 4) << 3) + r;
    size_t ob = (size_t)b * Ssz * Hsz + (size_t)qrow * Hsz + head * DHEAD;
#pragma unroll
    for (int nt = 0; nt < 4; ++nt)
      O[ob + nt * 16 + (lane & 15)] = f2bf(o[nt][r] * inv);
  }
}

// ---------------------------------------------------------------------------
// Elementwise / reduction helper kernels (bandwidth-trivial at 23.3 TB/s)
// ---------------------------------------------------------------------------
__global__ __launch_bounds__(256) void cast_transpose_f32_bf16(
    const float* __restrict__ W, __bf16* __restrict__ Wt, int K, int N) {
  int idx = blockIdx.x * 256 + threadIdx.x;   // Wt is [N,K]
  if (idx >= K * N) return;
  int n = idx / K, k = idx - n * K;
  Wt[idx] = f2bf(W[(size_t)k * N + n]);
}

__global__ __launch_bounds__(256) void cast_f32_bf16_strided(
    const float* __restrict__ X, __bf16* __restrict__ Y, int rows, int cols,
    int ldo) {
  int idx = blockIdx.x * 256 + threadIdx.x;
  if (idx >= rows * cols) return;
  int r = idx / cols, c = idx - r * cols;
  Y[(size_t)r * ldo + c] = f2bf(X[idx]);
}

__global__ __launch_bounds__(256) void softmax_rows_512(
    const float* __restrict__ in, __bf16* __restrict__ out) {
  int wave = threadIdx.x >> 5, lane = threadIdx.x & 31;
  int row = blockIdx.x * 8 + wave;
  const float* p = in + (size_t)row * Rsz;
  float v[16], mx = -3.0e38f;
#pragma unroll
  for (int i = 0; i < 16; ++i) { v[i] = p[lane + i * 32]; mx = fmaxf(mx, v[i]); }
#pragma unroll
  for (int off = 16; off >= 1; off >>= 1) mx = fmaxf(mx, __shfl_xor(mx, off));
  float sum = 0.f;
#pragma unroll
  for (int i = 0; i < 16; ++i) { v[i] = __expf(v[i] - mx); sum += v[i]; }
#pragma unroll
  for (int off = 16; off >= 1; off >>= 1) sum += __shfl_xor(sum, off);
  float inv = 1.f / sum;
  __bf16* q = out + (size_t)row * Rsz;
#pragma unroll
  for (int i = 0; i < 16; ++i) q[lane + i * 32] = f2bf(v[i] * inv);
}

__global__ __launch_bounds__(256) void ln_gelu_rows(
    const float* __restrict__ in, const float* __restrict__ g,
    const float* __restrict__ bt, __bf16* __restrict__ out) {
  int wave = threadIdx.x >> 5, lane = threadIdx.x & 31;
  int row = blockIdx.x * 8 + wave;
  const float* p = in + (size_t)row * Hsz;
  float v[32], sum = 0.f;
#pragma unroll
  for (int i = 0; i < 32; ++i) { v[i] = p[lane + i * 32]; sum += v[i]; }
#pragma unroll
  for (int off = 16; off >= 1; off >>= 1) sum += __shfl_xor(sum, off);
  float mean = sum * (1.f / Hsz);
  float var = 0.f;
#pragma unroll
  for (int i = 0; i < 32; ++i) { float d = v[i] - mean; var += d * d; }
#pragma unroll
  for (int off = 16; off >= 1; off >>= 1) var += __shfl_xor(var, off);
  float rstd = rsqrtf(var * (1.f / Hsz) + 1e-5f);
  __bf16* q = out + (size_t)row * Hsz;
#pragma unroll
  for (int i = 0; i < 32; ++i) {
    int c = lane + i * 32;
    float x = (v[i] - mean) * rstd * g[c] + bt[c];
    float u = 0.5f * x * (1.f + erff(x * 0.70710678118f));   // exact GELU
    q[c] = f2bf(u);
  }
}

__global__ __launch_bounds__(256) void confidence_rows(
    const float* __restrict__ h, const float* __restrict__ w,
    const float* __restrict__ bias, float* __restrict__ out) {
  int wave = threadIdx.x >> 5, lane = threadIdx.x & 31;
  int row = blockIdx.x * 8 + wave;
  const float* p = h + (size_t)row * Hsz;
  float s = 0.f;
#pragma unroll
  for (int i = 0; i < 32; ++i) s += p[lane + i * 32] * w[lane + i * 32];
#pragma unroll
  for (int off = 16; off >= 1; off >>= 1) s += __shfl_xor(s, off);
  if (lane == 0) out[row] = 1.f / (1.f + __expf(-(s + bias[0])));
}

// ---------------------------------------------------------------------------
// Orchestration
// ---------------------------------------------------------------------------
extern "C" void kernel_launch(void* const* d_in, const int* in_sizes, int n_in,
                              void* d_out, int out_size, void* d_ws,
                              size_t ws_size, hipStream_t stream) {
  (void)in_sizes; (void)n_in; (void)out_size; (void)ws_size;

  const float* hs        = (const float*)d_in[0];
  const float* rule_emb  = (const float*)d_in[1];
  const float* rsel_w    = (const float*)d_in[2];
  const float* rsel_b    = (const float*)d_in[3];
  const float* awq = (const float*)d_in[4];  const float* abq = (const float*)d_in[5];
  const float* awk = (const float*)d_in[6];  const float* abk = (const float*)d_in[7];
  const float* awv = (const float*)d_in[8];  const float* abv = (const float*)d_in[9];
  const float* awo = (const float*)d_in[10]; const float* abo = (const float*)d_in[11];
  const float* memory = (const float*)d_in[12];
  const float* mqw = (const float*)d_in[13]; const float* mqb = (const float*)d_in[14];
  const float* mkw = (const float*)d_in[15]; const float* mkb = (const float*)d_in[16];
  const float* mvw = (const float*)d_in[17]; const float* mvb = (const float*)d_in[18];
  const float* mawq = (const float*)d_in[19]; const float* mabq = (const float*)d_in[20];
  const float* mawk = (const float*)d_in[21]; const float* mabk = (const float*)d_in[22];
  const float* mawv = (const float*)d_in[23]; const float* mabv = (const float*)d_in[24];
  const float* mawo = (const float*)d_in[25]; const float* mabo = (const float*)d_in[26];
  const float* raw1 = (const float*)d_in[27]; const float* rab1 = (const float*)d_in[28];
  const float* lns  = (const float*)d_in[29]; const float* lnb  = (const float*)d_in[30];
  const float* raw2 = (const float*)d_in[31]; const float* rab2 = (const float*)d_in[32];
  const float* cw   = (const float*)d_in[33]; const float* cb   = (const float*)d_in[34];

  char* ws = (char*)d_ws;
  size_t off = 0;
  auto alloc = [&](size_t bytes) -> void* {
    void* p = ws + off;
    off += (bytes + 255) & ~(size_t)255;
    return p;
  };
  const size_t TOK = (size_t)Bsz * Ssz;            // 4096 rows

  // bf16 transposed weights (N x K)
  __bf16* rs_t  = (__bf16*)alloc((size_t)Rsz * Hsz * 2);
  __bf16* re_t  = (__bf16*)alloc((size_t)Hsz * Rsz * 2);
  __bf16* wq_t  = (__bf16*)alloc((size_t)Hsz * Hsz * 2);
  __bf16* wk_t  = (__bf16*)alloc((size_t)Hsz * Hsz * 2);
  __bf16* wv_t  = (__bf16*)alloc((size_t)Hsz * Hsz * 2);
  __bf16* wo_t  = (__bf16*)alloc((size_t)Hsz * Hsz * 2);
  __bf16* mq_t  = (__bf16*)alloc((size_t)Hsz * Hsz * 2);
  __bf16* mk_t  = (__bf16*)alloc((size_t)Hsz * Hsz * 2);
  __bf16* mv_t  = (__bf16*)alloc((size_t)Hsz * Hsz * 2);
  __bf16* maq_t = (__bf16*)alloc((size_t)Hsz * Hsz * 2);
  __bf16* mak_t = (__bf16*)alloc((size_t)Hsz * Hsz * 2);
  __bf16* mav_t = (__bf16*)alloc((size_t)Hsz * Hsz * 2);
  __bf16* mao_t = (__bf16*)alloc((size_t)Hsz * Hsz * 2);
  __bf16* ra1_t = (__bf16*)alloc((size_t)Hsz * 2 * Hsz * 2);
  __bf16* ra2_t = (__bf16*)alloc((size_t)Hsz * Hsz * 2);
  // activations
  __bf16* mem_bf = (__bf16*)alloc((size_t)Msz * Hsz * 2);
  __bf16* mk2    = (__bf16*)alloc((size_t)Msz * Hsz * 2);
  __bf16* mv2    = (__bf16*)alloc((size_t)Msz * Hsz * 2);
  __bf16* xcat   = (__bf16*)alloc(TOK * 2 * Hsz * 2);  // [h_bf | selected]
  __bf16* probs  = (__bf16*)alloc(TOK * Rsz * 2);
  __bf16* qb     = (__bf16*)alloc(TOK * Hsz * 2);
  __bf16* kb     = (__bf16*)alloc(TOK * Hsz * 2);
  __bf16* vb     = (__bf16*)alloc(TOK * Hsz * 2);
  __bf16* aout   = (__bf16*)alloc(TOK * Hsz * 2);
  float*  hbuf   = (float*)alloc(TOK * Hsz * 4);
  float*  logits = (float*)alloc(TOK * Rsz * 4);
  float*  upre   = (float*)alloc(TOK * Hsz * 4);

  auto tr = [&](const float* W, __bf16* Wt, int K, int N) {
    int total = K * N;
    cast_transpose_f32_bf16<<<(total + 255) / 256, 256, 0, stream>>>(W, Wt, K,
                                                                     N);
  };
  auto gemmB = [&](const __bf16* A, int lda, const __bf16* Bt, int ldb,
                   const float* bias, __bf16* outB, int ldob, int Md, int Nd,
                   int Kd) {
    dim3 g(Nd / BN, Md / BM);
    gemm_bf16_wmma<false, true, false><<<g, 256, 0, stream>>>(
        A, lda, Bt, ldb, bias, nullptr, 0, nullptr, 0, outB, ldob, Kd);
  };
  auto gemmF = [&](const __bf16* A, int lda, const __bf16* Bt, int ldb,
                   const float* bias, float* outF, int ldof, int Md, int Nd,
                   int Kd) {
    dim3 g(Nd / BN, Md / BM);
    gemm_bf16_wmma<true, false, false><<<g, 256, 0, stream>>>(
        A, lda, Bt, ldb, bias, nullptr, 0, outF, ldof, nullptr, 0, Kd);
  };
  auto gemmRes = [&](const __bf16* A, int lda, const __bf16* Bt,
                     const float* bias, int Kd) {
    // h = h + A@W + bias ; write f32 h and bf16 mirror into xcat[:, :H]
    dim3 g(Hsz / BN, (int)(TOK / BM));
    gemm_bf16_wmma<true, true, true><<<g, 256, 0, stream>>>(
        A, lda, Bt, Hsz, bias, hbuf, Hsz, hbuf, Hsz, xcat, 2 * Hsz, Kd);
  };
  auto flash = [&](const __bf16* Qp, const __bf16* Kq, const __bf16* Vq,
                   __bf16* Op, long kvbs) {
    dim3 g(Ssz / 128, NHEADS, Bsz);
    flash_attn_bf16<<<g, 256, 0, stream>>>(Qp, Kq, Vq, Op, (long)Ssz * Hsz,
                                           kvbs, 1024, Hsz, 0.125f);
  };

  // ---- one-time prep: weight transpose/cast, h init, memory projections ----
  tr(rsel_w, rs_t, Hsz, Rsz);
  tr(rule_emb, re_t, Rsz, Hsz);
  tr(awq, wq_t, Hsz, Hsz);  tr(awk, wk_t, Hsz, Hsz);
  tr(awv, wv_t, Hsz, Hsz);  tr(awo, wo_t, Hsz, Hsz);
  tr(mqw, mq_t, Hsz, Hsz);  tr(mkw, mk_t, Hsz, Hsz);  tr(mvw, mv_t, Hsz, Hsz);
  tr(mawq, maq_t, Hsz, Hsz); tr(mawk, mak_t, Hsz, Hsz);
  tr(mawv, mav_t, Hsz, Hsz); tr(mawo, mao_t, Hsz, Hsz);
  tr(raw1, ra1_t, 2 * Hsz, Hsz);
  tr(raw2, ra2_t, Hsz, Hsz);
  {
    int tot = Msz * Hsz;
    cast_f32_bf16_strided<<<(tot + 255) / 256, 256, 0, stream>>>(memory, mem_bf,
                                                                 Msz, Hsz, Hsz);
    int tot2 = (int)(TOK * Hsz);
    cast_f32_bf16_strided<<<(tot2 + 255) / 256, 256, 0, stream>>>(
        hs, xcat, (int)TOK, Hsz, 2 * Hsz);
  }
  hipMemcpyAsync(hbuf, hs, TOK * Hsz * sizeof(float), hipMemcpyDeviceToDevice,
                 stream);

  // mk2 = (memory@mem_k_w + b) @ mattn_wk + b ; mv2 likewise (step-invariant)
  gemmB(mem_bf, Hsz, mk_t, Hsz, mkb, qb, Hsz, Msz, Hsz, Hsz);
  gemmB(qb, Hsz, mak_t, Hsz, mabk, mk2, Hsz, Msz, Hsz, Hsz);
  gemmB(mem_bf, Hsz, mv_t, Hsz, mvb, qb, Hsz, Msz, Hsz, Hsz);
  gemmB(qb, Hsz, mav_t, Hsz, mabv, mv2, Hsz, Msz, Hsz, Hsz);

  // ---- reasoning steps ----
  for (int step = 0; step < NSTEPS; ++step) {
    // rule routing
    gemmF(xcat, 2 * Hsz, rs_t, Hsz, rsel_b, logits, Rsz, (int)TOK, Rsz, Hsz);
    softmax_rows_512<<<(int)(TOK / 8), 256, 0, stream>>>(logits, probs);
    gemmB(probs, Rsz, re_t, Rsz, nullptr, xcat + Hsz, 2 * Hsz, (int)TOK, Hsz,
          Rsz);  // selected -> second half of concat buffer

    // self-attention
    gemmB(xcat, 2 * Hsz, wq_t, Hsz, abq, qb, Hsz, (int)TOK, Hsz, Hsz);
    gemmB(xcat, 2 * Hsz, wk_t, Hsz, abk, kb, Hsz, (int)TOK, Hsz, Hsz);
    gemmB(xcat, 2 * Hsz, wv_t, Hsz, abv, vb, Hsz, (int)TOK, Hsz, Hsz);
    flash(qb, kb, vb, aout, (long)Ssz * Hsz);
    gemmRes(aout, Hsz, wo_t, abo, Hsz);  // h += attn_out @ wo + bo

    // memory attention (k/v precomputed, shared across batch)
    gemmB(xcat, 2 * Hsz, mq_t, Hsz, mqb, qb, Hsz, (int)TOK, Hsz, Hsz);   // mq
    gemmB(qb, Hsz, maq_t, Hsz, mabq, kb, Hsz, (int)TOK, Hsz, Hsz);       // qh
    flash(kb, mk2, mv2, aout, 0);
    gemmRes(aout, Hsz, mao_t, mabo, Hsz);  // h += mattn_out @ wo + bo

    // rule application MLP on [h ; selected]
    gemmF(xcat, 2 * Hsz, ra1_t, 2 * Hsz, rab1, upre, Hsz, (int)TOK, Hsz,
          2 * Hsz);
    ln_gelu_rows<<<(int)(TOK / 8), 256, 0, stream>>>(upre, lns, lnb, vb);
    gemmRes(vb, Hsz, ra2_t, rab2, Hsz);  // h += u @ ra_w2 + b2
  }

  // ---- outputs: h (f32) then confidence (f32) ----
  hipMemcpyAsync(d_out, hbuf, TOK * Hsz * sizeof(float),
                 hipMemcpyDeviceToDevice, stream);
  confidence_rows<<<(int)(TOK / 8), 256, 0, stream>>>(
      hbuf, cw, cb, (float*)d_out + TOK * Hsz);
}
